// GQABlock_14568529068626
// MI455X (gfx1250) — matrix operand order repaired and checked by reading, not compiled, not run on hardware
//
#include <hip/hip_runtime.h>

// ---------------------------------------------------------------------------
// GQA block for gfx1250 (MI455X), bf16 WMMA pipeline:
//   x,W -> bf16 ; Q=xWq^T*0.125 ; K=xWk^T ; Vt=transpose(xWv^T) ;
//   flash attention (causal, GQA kv = hq%8) ; out = ctx Wo^T (f32)
// GEMMs: workgroup-cooperative 128x128 tiles, A/B double-buffer-staged to LDS
// with GLOBAL_LOAD_ASYNC_TO_LDS_B128 (ASYNCcnt), consumed by
// v_wmma_f32_16x16x32_bf16. Attention: 4-wave workgroup / 64 query rows;
// K/V tiles double-buffer async-staged to LDS and shared by all waves.
// ---------------------------------------------------------------------------

#define B_   2
#define S_   2048
#define E_   2048
#define HQ_  32
#define HKV_ 8
#define D_   64

typedef __attribute__((ext_vector_type(16))) __bf16 v16bf;
typedef __attribute__((ext_vector_type(8)))  float  v8f;

union BfTile { v16bf v; uint4 q[2]; };

__device__ __forceinline__ uint4 ld4(const unsigned short* p) {
  return *(const uint4*)p;
}

__device__ __forceinline__ unsigned short f2bf(float f) {
  unsigned int u = __float_as_uint(f);
  unsigned int r = u + 0x7FFFu + ((u >> 16) & 1u);   // round-to-nearest-even
  return (unsigned short)(r >> 16);
}

__device__ __forceinline__ v8f wmma_bf16(const BfTile& a, const BfTile& b, v8f c) {
  return __builtin_amdgcn_wmma_f32_16x16x32_bf16(
      /*neg_a=*/false, a.v, /*neg_b=*/false, b.v,
      /*c_mod=*/(short)0, c, /*reuse_a=*/false, /*reuse_b=*/false);
}

__device__ __forceinline__ float rowmax16(float v) {
#pragma unroll
  for (int m = 1; m < 16; m <<= 1) v = fmaxf(v, __shfl_xor(v, m, 32));
  return v;
}
__device__ __forceinline__ float rowsum16(float v) {
#pragma unroll
  for (int m = 1; m < 16; m <<= 1) v += __shfl_xor(v, m, 32);
  return v;
}

// Async global->LDS copies (ASYNCcnt tracked). IOFFSET is added to BOTH the
// LDS and the global byte address (ISA 15.18.3), so consecutive 16B chunks
// stay contiguous on both sides.
__device__ __forceinline__ void async_copy64(unsigned ldsOff, unsigned gOff,
                                             const void* base) {
  asm volatile(
      "global_load_async_to_lds_b128 %0, %1, %2 offset:0\n\t"
      "global_load_async_to_lds_b128 %0, %1, %2 offset:16\n\t"
      "global_load_async_to_lds_b128 %0, %1, %2 offset:32\n\t"
      "global_load_async_to_lds_b128 %0, %1, %2 offset:48"
      :: "v"(ldsOff), "v"(gOff), "s"((unsigned long long)base)
      : "memory");
}
__device__ __forceinline__ void async_copy32(unsigned ldsOff, unsigned gOff,
                                             const void* base) {
  asm volatile(
      "global_load_async_to_lds_b128 %0, %1, %2 offset:0\n\t"
      "global_load_async_to_lds_b128 %0, %1, %2 offset:16"
      :: "v"(ldsOff), "v"(gOff), "s"((unsigned long long)base)
      : "memory");
}
// Wait until this wave's outstanding async ops <= N (they complete in order,
// so "<= count just issued" means everything older has landed).
template <int N>
__device__ __forceinline__ void wait_async() {
  asm volatile("s_wait_asynccnt %0" :: "n"(N) : "memory");
}

// Truncate a generic pointer to its LDS byte offset (flat-LDS keeps it in [31:0]).
__device__ __forceinline__ unsigned lds_off(const void* p) {
  return (unsigned)(unsigned long long)p;
}

// ---------------------------------------------------------------------------
// f32 -> bf16 conversion
// ---------------------------------------------------------------------------
__global__ void cvt_bf16_kernel(const float* __restrict__ in,
                                unsigned short* __restrict__ out, int n) {
  int i = blockIdx.x * blockDim.x + threadIdx.x;
  if (i < n) out[i] = f2bf(in[i]);
}

// ---------------------------------------------------------------------------
// Y[M,N] = A[M,K] * W[N,K]^T ; workgroup (128 thr = 4 waves) does a 128x128
// output tile; each wave owns a 64x64 quadrant (16 WMMA accumulators).
// K-step(32): A/B 128x32 bf16 slices double-buffer async-staged into LDS
// (80B row pitch avoids bank conflicts on the 16B operand reads); tile i+1's
// DMA overlaps tile i's WMMAs.
// A-operand layout: lane = row m, lanes<16 hold K {0..7,16..23}, lanes>=16 {8..15,24..31}
// B-operand layout: lane = col n, lanes<16 hold K {0..7,16..23}, lanes>=16 {8..15,24..31}
// MODE 0: bf16 out row-major *scale | MODE 1: f32 out | MODE 2: Vt transposed store
// ---------------------------------------------------------------------------
template <int MODE>
__global__ void gemm_coop_kernel(const unsigned short* __restrict__ A,
                                 const unsigned short* __restrict__ W,
                                 void* __restrict__ outp,
                                 int N, int K, float scale) {
  __shared__ unsigned short ldsA[2][128 * 40];
  __shared__ unsigned short ldsB[2][128 * 40];

  const int tid  = threadIdx.x;
  const int wave = __builtin_amdgcn_readfirstlane(tid) >> 5;  // scalar wave id
  const int lane = tid & 31;
  const int col  = lane & 15;
  const int hi   = lane >> 4;
  const int wm   = wave >> 1;          // wave quadrant row (0..1)
  const int wn   = wave & 1;           // wave quadrant col (0..1)
  const int nBase = blockIdx.x * 128;
  const int mBase = blockIdx.y * 128;

  const unsigned aLds[2] = {lds_off(&ldsA[0][tid * 40]), lds_off(&ldsA[1][tid * 40])};
  const unsigned bLds[2] = {lds_off(&ldsB[0][tid * 40]), lds_off(&ldsB[1][tid * 40])};
  const size_t aRow = (size_t)(mBase + tid) * K;   // this thread stages A row mBase+tid
  const size_t bRow = (size_t)(nBase + tid) * K;   // and W row nBase+tid

  v8f acc[4][4] = {};

  // prologue: stage k-slice 0 into buffer 0 (8 async ops per wave)
  async_copy64(aLds[0], (unsigned)(aRow * 2), A);
  async_copy64(bLds[0], (unsigned)(bRow * 2), W);

  const int steps = K >> 5;
  for (int st = 0; st < steps; ++st) {
    const int cur = st & 1;
    if (st + 1 < steps) {
      const int k1 = (st + 1) << 5;
      async_copy64(aLds[cur ^ 1], (unsigned)((aRow + k1) * 2), A);
      async_copy64(bLds[cur ^ 1], (unsigned)((bRow + k1) * 2), W);
      wait_async<8>();   // current slice landed; next slice still in flight
    } else {
      wait_async<0>();
    }
    __syncthreads();

    const unsigned short* aBuf = ldsA[cur];
    const unsigned short* bBuf = ldsB[cur];
    BfTile a[4];
#pragma unroll
    for (int mi = 0; mi < 4; ++mi) {
      const unsigned short* ar = aBuf + (wm * 64 + mi * 16 + col) * 40;
      a[mi].q[0] = *(const uint4*)(ar + hi * 8);
      a[mi].q[1] = *(const uint4*)(ar + hi * 8 + 16);
    }
#pragma unroll
    for (int nj = 0; nj < 4; ++nj) {
      const unsigned short* br = bBuf + (wn * 64 + nj * 16 + col) * 40;
      BfTile b;
      b.q[0] = *(const uint4*)(br + hi * 8);
      b.q[1] = *(const uint4*)(br + hi * 8 + 16);
#pragma unroll
      for (int mi = 0; mi < 4; ++mi)
        acc[mi][nj] = wmma_bf16(a[mi], b, acc[mi][nj]);
    }
    __syncthreads();   // all waves done reading before this buffer is restaged
  }

#pragma unroll
  for (int mi = 0; mi < 4; ++mi) {
#pragma unroll
    for (int nj = 0; nj < 4; ++nj) {
#pragma unroll
      for (int r = 0; r < 8; ++r) {
        const int row  = mBase + wm * 64 + mi * 16 + r + 8 * hi;
        const int coln = nBase + wn * 64 + nj * 16 + col;
        const float v  = acc[mi][nj][r] * scale;
        if (MODE == 0) {
          ((unsigned short*)outp)[(size_t)row * N + coln] = f2bf(v);
        } else if (MODE == 1) {
          ((float*)outp)[(size_t)row * N + coln] = v;
        } else {  // MODE 2: V transposed store Vt[(b*8+h)*64+dd][s]
          const int b  = row >> 11;       // row = b*S + s
          const int s  = row & (S_ - 1);
          const int h  = coln >> 6;       // coln = h*64 + dd
          const int dd = coln & 63;
          ((unsigned short*)outp)[(((size_t)(b * HKV_ + h) * D_ + dd) << 11) + s] = f2bf(v);
        }
      }
    }
  }
}

// ---------------------------------------------------------------------------
// Flash attention, causal, GQA (kv head = hq % 8).
// grid = (S/64, HQ, B), 128 threads (4 waves). Workgroup covers 64 query rows;
// wave w owns rows sw0 = s0+16w .. +15 (16 rows -> no register spills).
// Per 32-wide T-tile: K (32x64) and Vt (64x32) double-buffer async-staged into
// LDS and consumed by all 4 waves. Causal masking only on diagonal tiles;
// wave-uniform (scalar) participation branches keep EXEC all-ones at WMMAs.
//   qbf: [B,S,E] bf16 (pre-scaled by 1/8), kbf: [B,S,512] bf16,
//   vt : [B,HKV,64,S] bf16, ctx out: [B,S,E] bf16 (feature = hq*64+dd)
// ---------------------------------------------------------------------------
__global__ void attn_kernel(const unsigned short* __restrict__ qbf,
                            const unsigned short* __restrict__ kbf,
                            const unsigned short* __restrict__ vt,
                            unsigned short* __restrict__ ctx) {
  __shared__ unsigned short ldsK[2][32 * 72];   // K tile: 32 t-rows x 64 k, pitch 72
  __shared__ unsigned short ldsV[2][64 * 40];   // Vt tile: 64 d-rows x 32 t, pitch 40
  __shared__ unsigned short ldsP[4][16 * 40];   // per-wave 16x32 P tile

  const int tid  = threadIdx.x;
  const int w    = __builtin_amdgcn_readfirstlane(tid) >> 5;  // scalar wave id
  const int lane = tid & 31;
  const int col  = lane & 15;
  const int hi   = lane >> 4;
  const int s0   = blockIdx.x * 64;
  const int sw0  = s0 + w * 16;        // this wave's first query row (scalar)
  const int hq   = blockIdx.y;
  const int b    = blockIdx.z;
  const int g    = hq & (HKV_ - 1);

  // Q tile for this wave (rows sw0..sw0+15), K = d = 64 -> two K=32 chunks
  const unsigned short* qrow =
      qbf + (size_t)(b * S_ + sw0 + col) * E_ + hq * D_;
  BfTile aq0, aq1;
  aq0.q[0] = ld4(qrow + hi * 8);
  aq0.q[1] = ld4(qrow + hi * 8 + 16);
  aq1.q[0] = ld4(qrow + 32 + hi * 8);
  aq1.q[1] = ld4(qrow + 32 + hi * 8 + 16);

  // staging assignments (whole workgroup)
  const int ktRow = tid >> 2, ktPart = tid & 3;   // K: 32 rows, 4 thr/row, 32B each
  const int vRow  = tid >> 1, vPart  = tid & 1;   // V: 64 rows, 2 thr/row, 32B each
  const unsigned kLds[2] = {lds_off(&ldsK[0][ktRow * 72]) + (unsigned)ktPart * 32,
                            lds_off(&ldsK[1][ktRow * 72]) + (unsigned)ktPart * 32};
  const unsigned vLds[2] = {lds_off(&ldsV[0][vRow * 40]) + (unsigned)vPart * 32,
                            lds_off(&ldsV[1][vRow * 40]) + (unsigned)vPart * 32};
  const size_t kRowBase = ((size_t)(b * S_) * (HKV_ * D_)) + g * D_;
  const size_t vRowBase = ((size_t)(b * HKV_ + g) * D_ + vRow) * S_;

  float mr[8], ls[8];
#pragma unroll
  for (int r = 0; r < 8; ++r) { mr[r] = -1e30f; ls[r] = 0.0f; }
  v8f acc[4] = {};

  const int nSteps = ((s0 + 63) >> 5) + 1;   // T-tiles t0 = st*32

  // prologue: stage tile 0 into buffer 0 (4 async ops per wave)
  async_copy32(kLds[0],
               (unsigned)((kRowBase + (size_t)ktRow * (HKV_ * D_)) * 2 + ktPart * 32),
               kbf);
  async_copy32(vLds[0], (unsigned)((vRowBase + vPart * 16) * 2), vt);

  for (int st = 0; st < nSteps; ++st) {
    const int t0  = st << 5;
    const int cur = st & 1;
    if (st + 1 < nSteps) {
      const int t1 = t0 + 32;
      async_copy32(kLds[cur ^ 1],
                   (unsigned)((kRowBase + (size_t)(t1 + ktRow) * (HKV_ * D_)) * 2 +
                              ktPart * 32),
                   kbf);
      async_copy32(vLds[cur ^ 1], (unsigned)((vRowBase + t1 + vPart * 16) * 2), vt);
      wait_async<4>();   // current tile landed; next tile still in flight
    } else {
      wait_async<0>();
    }
    __syncthreads();

    if (t0 <= sw0 + 15) {  // scalar: wave has at least one unmasked column
      const unsigned short* kBuf = ldsK[cur];
      const unsigned short* vBuf = ldsV[cur];

      // ---- scores: 16x32 tile as two 16x16 subtiles, K=64 ----
      v8f sc[2];
#pragma unroll
      for (int j = 0; j < 2; ++j) {
        const unsigned short* kr = kBuf + (j * 16 + col) * 72;
        BfTile b0, b1;
        b0.q[0] = *(const uint4*)(kr + hi * 8);
        b0.q[1] = *(const uint4*)(kr + hi * 8 + 16);
        b1.q[0] = *(const uint4*)(kr + 32 + hi * 8);
        b1.q[1] = *(const uint4*)(kr + 32 + hi * 8 + 16);
        v8f z = {};
        z = wmma_bf16(aq0, b0, z);
        z = wmma_bf16(aq1, b1, z);
        sc[j] = z;
      }

      // ---- causal mask, only on diagonal tiles (scalar condition) ----
      if (t0 + 31 > sw0) {
#pragma unroll
        for (int j = 0; j < 2; ++j) {
          const int tc = t0 + j * 16 + col;
#pragma unroll
          for (int r = 0; r < 8; ++r)
            if (tc > sw0 + r + 8 * hi) sc[j][r] = -1e30f;
        }
      }

      // ---- online softmax ----
      float esc[8];
#pragma unroll
      for (int r = 0; r < 8; ++r) {
        float tm = rowmax16(fmaxf(sc[0][r], sc[1][r]));
        float nm = fmaxf(mr[r], tm);
        esc[r]   = __expf(mr[r] - nm);
        float p0 = __expf(sc[0][r] - nm);
        float p1 = __expf(sc[1][r] - nm);
        sc[0][r] = p0;
        sc[1][r] = p1;
        ls[r] = ls[r] * esc[r] + rowsum16(p0 + p1);
        mr[r] = nm;
      }
#pragma unroll
      for (int nj = 0; nj < 4; ++nj)
#pragma unroll
        for (int r = 0; r < 8; ++r) acc[nj][r] *= esc[r];

      // ---- stage P (wave-private; same-wave LDS ops are in order) ----
#pragma unroll
      for (int j = 0; j < 2; ++j)
#pragma unroll
        for (int r = 0; r < 8; ++r)
          ldsP[w][(r + 8 * hi) * 40 + j * 16 + col] = f2bf(sc[j][r]);

      BfTile ap;
      {
        const unsigned short* pr = &ldsP[w][col * 40];
        ap.q[0] = *(const uint4*)(pr + hi * 8);
        ap.q[1] = *(const uint4*)(pr + hi * 8 + 16);
      }

      // ---- context: acc += P (16x32) * V (32x64) from LDS Vt tile ----
#pragma unroll
      for (int nj = 0; nj < 4; ++nj) {
        const unsigned short* vr = vBuf + (nj * 16 + col) * 40;
        BfTile bv;
        bv.q[0] = *(const uint4*)(vr + hi * 8);
        bv.q[1] = *(const uint4*)(vr + hi * 8 + 16);
        acc[nj] = wmma_bf16(ap, bv, acc[nj]);
      }
    }
    __syncthreads();  // all waves done reading before this buffer is restaged
  }

  // ---- normalize and store context, feature index = hq*64 + dd ----
#pragma unroll
  for (int nj = 0; nj < 4; ++nj)
#pragma unroll
    for (int r = 0; r < 8; ++r) {
      const float v  = acc[nj][r] / ls[r];
      const int srow = sw0 + r + 8 * hi;
      ctx[(size_t)(b * S_ + srow) * E_ + hq * D_ + nj * 16 + col] = f2bf(v);
    }
}

// ---------------------------------------------------------------------------
// Workspace layout (bytes). ctx aliases xbf (dead after projections). ~60 MB.
// ---------------------------------------------------------------------------
#define OFF_XBF 0u                 /* 4096*2048*2 = 16,777,216 ; also ctx */
#define OFF_WQ  16777216u          /* 2048*2048*2 =  8,388,608 */
#define OFF_WK  25165824u          /*  512*2048*2 =  2,097,152 */
#define OFF_WV  27262976u          /*  512*2048*2 =  2,097,152 */
#define OFF_WO  29360128u          /* 2048*2048*2 =  8,388,608 */
#define OFF_Q   37748736u          /* 4096*2048*2 = 16,777,216 */
#define OFF_K   54525952u          /* 4096* 512*2 =  4,194,304 */
#define OFF_VT  58720256u          /* 4096* 512*2 =  4,194,304 */

extern "C" void kernel_launch(void* const* d_in, const int* in_sizes, int n_in,
                              void* d_out, int out_size, void* d_ws, size_t ws_size,
                              hipStream_t stream) {
  const float* x  = (const float*)d_in[0];
  // d_in[1] = attention_mask (causal, reproduced analytically in-kernel)
  const float* Wq = (const float*)d_in[2];
  const float* Wk = (const float*)d_in[3];
  const float* Wv = (const float*)d_in[4];
  const float* Wo = (const float*)d_in[5];

  char* ws = (char*)d_ws;
  unsigned short* xbf = (unsigned short*)(ws + OFF_XBF);
  unsigned short* wqb = (unsigned short*)(ws + OFF_WQ);
  unsigned short* wkb = (unsigned short*)(ws + OFF_WK);
  unsigned short* wvb = (unsigned short*)(ws + OFF_WV);
  unsigned short* wob = (unsigned short*)(ws + OFF_WO);
  unsigned short* qb  = (unsigned short*)(ws + OFF_Q);
  unsigned short* kb  = (unsigned short*)(ws + OFF_K);
  unsigned short* vtb = (unsigned short*)(ws + OFF_VT);
  unsigned short* ctx = (unsigned short*)(ws + OFF_XBF);  // alias

  const int M = B_ * S_;  // 4096

  // f32 -> bf16 conversions
  {
    int n;
    n = M * E_;        cvt_bf16_kernel<<<(n + 255) / 256, 256, 0, stream>>>(x,  xbf, n);
    n = E_ * E_;       cvt_bf16_kernel<<<(n + 255) / 256, 256, 0, stream>>>(Wq, wqb, n);
    n = HKV_*D_ * E_;  cvt_bf16_kernel<<<(n + 255) / 256, 256, 0, stream>>>(Wk, wkb, n);
    n = HKV_*D_ * E_;  cvt_bf16_kernel<<<(n + 255) / 256, 256, 0, stream>>>(Wv, wvb, n);
    n = E_ * E_;       cvt_bf16_kernel<<<(n + 255) / 256, 256, 0, stream>>>(Wo, wob, n);
  }

  // Projections: Q (scale folds 1/sqrt(d)=0.125), K, V (transposed store)
  gemm_coop_kernel<0><<<dim3(E_ / 128, M / 128), 128, 0, stream>>>(xbf, wqb, qb, E_, E_, 0.125f);
  gemm_coop_kernel<0><<<dim3((HKV_*D_) / 128, M / 128), 128, 0, stream>>>(xbf, wkb, kb, HKV_*D_, E_, 1.0f);
  gemm_coop_kernel<2><<<dim3((HKV_*D_) / 128, M / 128), 128, 0, stream>>>(xbf, wvb, vtb, HKV_*D_, E_, 1.0f);

  // Flash attention (64 query rows per 4-wave workgroup, shared K/V tiles)
  attn_kernel<<<dim3(S_ / 64, HQ_, B_), 128, 0, stream>>>(qb, kb, vtb, ctx);

  // Output projection (f32 out)
  gemm_coop_kernel<1><<<dim3(E_ / 128, M / 128), 128, 0, stream>>>(ctx, wob, d_out, E_, E_, 1.0f);
}
